// Vikl_Loss_Adaptive_HRW_71047349011043
// MI455X (gfx1250) — matrix operand order repaired and checked by reading, not compiled, not run on hardware
//
#include <hip/hip_runtime.h>

// ---------------------------------------------------------------------------
// Problem constants (from reference)
// ---------------------------------------------------------------------------
#define BB        4096
#define DD        128
#define TAU_PLUS  0.1f
#define TEMP      0.5f
#define INV_TEMP  2.0f        // 1/TEMP
#define NCHUNK    8           // column chunks per row-block (no atomics needed)
#define WAVES_PER_WG 8        // 8 row tiles per workgroup share staged B tiles

// ---------------------------------------------------------------------------
// WMMA types (gfx1250, wave32)
// ---------------------------------------------------------------------------
typedef __bf16        v16bf __attribute__((ext_vector_type(16)));
typedef float         v8f   __attribute__((ext_vector_type(8)));
typedef unsigned int  v4u   __attribute__((ext_vector_type(4)));

union Frag {
    v16bf v;
    v4u   q[2];
};

// RNE float -> bf16 bits (manual, avoids __bf16 scalar arithmetic)
static __device__ __forceinline__ unsigned short f2bf(float x) {
    unsigned u = __float_as_uint(x);
    u += 0x7FFFu + ((u >> 16) & 1u);
    return (unsigned short)(u >> 16);
}
static __device__ __forceinline__ float bf2f(unsigned short h) {
    return __uint_as_float(((unsigned)h) << 16);
}

// Issue one 16-byte async global->LDS copy (CDNA5 path, tracked by ASYNCcnt).
static __device__ __forceinline__ void async_copy16(unsigned lds_addr, const void* gaddr) {
    asm volatile("global_load_async_to_lds_b128 %0, %1, off"
                 :: "v"(lds_addr), "v"(gaddr) : "memory");
}

// ---------------------------------------------------------------------------
// Kernel 1: build bf16 hi/lo split planes.
//   G1 = img1[idx], G2 = img2[idx], AT = attr[idx]   (4096 x 128 each)
//   C  = concat(img1, img2)                          (8192 x 128)
// ---------------------------------------------------------------------------
__global__ void prep_kernel(const float* __restrict__ img1,
                            const float* __restrict__ img2,
                            const float* __restrict__ attr,
                            const int*   __restrict__ idx,
                            unsigned short* __restrict__ G1hi, unsigned short* __restrict__ G1lo,
                            unsigned short* __restrict__ G2hi, unsigned short* __restrict__ G2lo,
                            unsigned short* __restrict__ AThi, unsigned short* __restrict__ ATlo,
                            unsigned short* __restrict__ Chi,  unsigned short* __restrict__ Clo) {
    int t = blockIdx.x * blockDim.x + threadIdx.x;
    if (t >= BB * DD) return;
    int row = t >> 7;
    int col = t & (DD - 1);
    int g   = idx[row];
    size_t goff = (size_t)g * DD + col;

    float x;
    unsigned short h;

    x = img1[goff]; h = f2bf(x);
    G1hi[t] = h; G1lo[t] = f2bf(x - bf2f(h));

    x = img2[goff]; h = f2bf(x);
    G2hi[t] = h; G2lo[t] = f2bf(x - bf2f(h));

    x = attr[goff]; h = f2bf(x);
    AThi[t] = h; ATlo[t] = f2bf(x - bf2f(h));

    // un-gathered concat for the intra loss
    x = img1[t]; h = f2bf(x);
    Chi[t] = h; Clo[t] = f2bf(x - bf2f(h));

    size_t t2 = (size_t)(row + BB) * DD + col;
    x = img2[t]; h = f2bf(x);
    Chi[t2] = h; Clo[t2] = f2bf(x - bf2f(h));
}

// ---------------------------------------------------------------------------
// Kernel 2: pos terms in full fp32 from the original inputs.
// ---------------------------------------------------------------------------
__global__ void pos_kernel(const float* __restrict__ img1,
                           const float* __restrict__ img2,
                           const float* __restrict__ attr,
                           const int*   __restrict__ idx,
                           float* __restrict__ p1, float* __restrict__ p2,
                           float* __restrict__ pI) {
    int i = blockIdx.x * blockDim.x + threadIdx.x;
    if (i >= BB) return;
    int g = idx[i];
    const float* x1g = img1 + (size_t)g * DD;
    const float* x2g = img2 + (size_t)g * DD;
    const float* ag  = attr + (size_t)g * DD;
    const float* x1  = img1 + (size_t)i * DD;
    const float* x2  = img2 + (size_t)i * DD;
    float d1 = 0.f, d2 = 0.f, d3 = 0.f;
#pragma unroll 16
    for (int k = 0; k < DD; ++k) {
        d1 = fmaf(x1g[k], ag[k], d1);
        d2 = fmaf(x2g[k], ag[k], d2);
        d3 = fmaf(x1[k],  x2[k], d3);
    }
    p1[i] = __expf(d1 * INV_TEMP);
    p2[i] = __expf(d2 * INV_TEMP);
    pI[i] = __expf(d3 * INV_TEMP);
}

// ---------------------------------------------------------------------------
// Kernel 3: WMMA row-sum kernel with async-LDS B staging.
//   sim = A @ B^T / TEMP  (A,B: nRows x 128, split bf16 hi/lo)
//   Workgroup = 8 waves owning 8 consecutive 16-row tiles x one column chunk.
//   B tiles (16 rows x 128 bf16, hi+lo = 8KB) are staged into double-buffered
//   LDS via global_load_async_to_lds_b128 and shared by all 8 waves, cutting
//   L2 read traffic 8x. Element (i,j) masked iff ((i^j) & maskAnd) == 0.
//   Per-(chunk,row) result slots -> deterministic, no atomics.
// ---------------------------------------------------------------------------
__global__ void __launch_bounds__(256)
gemm_rowsum_kernel(const unsigned short* __restrict__ Ahi,
                   const unsigned short* __restrict__ Alo,
                   const unsigned short* __restrict__ Bhi,
                   const unsigned short* __restrict__ Blo,
                   float* __restrict__ S1, float* __restrict__ S2,
                   int nRows, int chunkTiles, int maskAnd) {
    __shared__ __align__(16) unsigned short smem[2][2][16 * DD];  // [buf][hi/lo][16x128] = 16KB

    const int tid    = threadIdx.x;
    const int wave   = tid >> 5;
    const int lane   = tid & 31;
    const int lane15 = lane & 15;
    const int half   = lane >> 4;               // 0: lanes 0-15, 1: lanes 16-31

    const int rowSuper = blockIdx.x / NCHUNK;
    const int chunk    = blockIdx.x - rowSuper * NCHUNK;
    const int rowTile  = rowSuper * WAVES_PER_WG + wave;
    const int m0       = rowTile * 16;
    const int r        = m0 + lane15;           // A-matrix row owned by this lane

    // ---- Preload A fragments for all 4 K-steps (hi and lo planes) ----------
    // ISA 16-bit A 16x32 layout: lane m (half=0) holds K k0+0..7 (VGPR0-3) and
    // K k0+16..23 (VGPR4-7); lane m+16 (half=1) holds K k0+8..15 / k0+24..31.
    Frag ahi[4], alo[4];
#pragma unroll
    for (int t = 0; t < 4; ++t) {
        const size_t base = (size_t)r * DD + t * 32 + half * 8;
        ahi[t].q[0] = *(const v4u*)(Ahi + base);
        ahi[t].q[1] = *(const v4u*)(Ahi + base + 16);
        alo[t].q[0] = *(const v4u*)(Alo + base);
        alo[t].q[1] = *(const v4u*)(Alo + base + 16);
    }

    float s1acc[8], s2acc[8];
#pragma unroll
    for (int v = 0; v < 8; ++v) { s1acc[v] = 0.f; s2acc[v] = 0.f; }

    const int ctile0 = chunk * chunkTiles;

    // Stage first B tile into buffer 0: each of 256 threads copies 16B per
    // plane (16 rows x 256B = 4KB contiguous per plane).
    {
        const int c0 = ctile0 * 16;
        async_copy16((unsigned)(uintptr_t)((char*)&smem[0][0][0] + tid * 16),
                     (const char*)(Bhi + (size_t)c0 * DD) + tid * 16);
        async_copy16((unsigned)(uintptr_t)((char*)&smem[0][1][0] + tid * 16),
                     (const char*)(Blo + (size_t)c0 * DD) + tid * 16);
    }

    int cur = 0;
    for (int ct = 0; ct < chunkTiles; ++ct) {
        if (ct + 1 < chunkTiles) {
            // prefetch next tile into the other buffer (safe: trailing barrier
            // of the previous iteration guarantees nobody still reads it)
            const int c0n = (ctile0 + ct + 1) * 16;
            const int nb  = cur ^ 1;
            async_copy16((unsigned)(uintptr_t)((char*)&smem[nb][0][0] + tid * 16),
                         (const char*)(Bhi + (size_t)c0n * DD) + tid * 16);
            async_copy16((unsigned)(uintptr_t)((char*)&smem[nb][1][0] + tid * 16),
                         (const char*)(Blo + (size_t)c0n * DD) + tid * 16);
            // async loads complete in order: <=2 outstanding means the current
            // tile's 2 copies have landed in LDS
            asm volatile("s_wait_asynccnt 0x2" ::: "memory");
        } else {
            asm volatile("s_wait_asynccnt 0x0" ::: "memory");
        }
        __syncthreads();   // all waves' portions of the current tile are visible

        const unsigned short* shi = &smem[cur][0][0];
        const unsigned short* slo = &smem[cur][1][0];
        const int c0 = (ctile0 + ct) * 16;

        v8f c = {};
#pragma unroll
        for (int t = 0; t < 4; ++t) {
            // B 32x16 layout (matches A): lane n holds K k0+0..7 / k0+16..23,
            // lane n+16 holds K k0+8..15 / k0+24..31.  (ds_load_b128 x2 per fragment)
            const size_t boff = (size_t)lane15 * DD + t * 32 + half * 8;
            Frag bhi, blo;
            bhi.q[0] = *(const v4u*)(shi + boff);
            bhi.q[1] = *(const v4u*)(shi + boff + 16);
            blo.q[0] = *(const v4u*)(slo + boff);
            blo.q[1] = *(const v4u*)(slo + boff + 16);
            // split-precision product: hi*hi + hi*lo + lo*hi  (~fp32 accuracy)
            c = __builtin_amdgcn_wmma_f32_16x16x32_bf16(false, ahi[t].v, false, bhi.v,
                                                        (short)0, c, false, false);
            c = __builtin_amdgcn_wmma_f32_16x16x32_bf16(false, ahi[t].v, false, blo.v,
                                                        (short)0, c, false, false);
            c = __builtin_amdgcn_wmma_f32_16x16x32_bf16(false, alo[t].v, false, bhi.v,
                                                        (short)0, c, false, false);
        }

        // C layout: VGPR v holds row m0+v (lanes 0-15) / m0+v+8 (lanes 16-31),
        // column = c0 + lane15.
        const int colg = c0 + lane15;
#pragma unroll
        for (int v = 0; v < 8; ++v) {
            const int rowg = m0 + v + 8 * half;
            const float s  = c[v] * INV_TEMP;
            const float e  = (((rowg ^ colg) & maskAnd) == 0) ? 0.0f : __expf(s);
            s1acc[v] += e;
            s2acc[v] += e * e;   // exp(2*sim) since BETA == 1
        }

        __syncthreads();   // protect smem[cur] before next iteration's prefetch
        cur ^= 1;
    }

    // Reduce across the 16 lanes of each half, then store per (chunk,row).
#pragma unroll
    for (int v = 0; v < 8; ++v) {
        float a = s1acc[v], b = s2acc[v];
        for (int off = 8; off > 0; off >>= 1) {
            a += __shfl_xor(a, off, 32);
            b += __shfl_xor(b, off, 32);
        }
        if (lane15 == 0) {
            const int rowg = m0 + v + 8 * half;
            S1[(size_t)chunk * nRows + rowg] = a;
            S2[(size_t)chunk * nRows + rowg] = b;
        }
    }
}

// ---------------------------------------------------------------------------
// Kernel 4: finalize (hard-negative formula + means + combine)
// ---------------------------------------------------------------------------
static __device__ __forceinline__ float loss_term(float s1, float s2, float pos, float nf) {
    const float rw = (s2 * nf) / s1;
    float ng = (rw - TAU_PLUS * nf * pos) * (1.0f / (1.0f - TAU_PLUS));
    const float fl = nf * __expf(-1.0f / TEMP);
    ng = fmaxf(ng, fl);
    return __logf((pos + ng) / pos);   // == -log(pos / (pos + ng))
}

__global__ void finalize_kernel(const float* __restrict__ S1a, const float* __restrict__ S2a,
                                const float* __restrict__ S1b, const float* __restrict__ S2b,
                                const float* __restrict__ S1c, const float* __restrict__ S2c,
                                const float* __restrict__ p1,  const float* __restrict__ p2,
                                const float* __restrict__ pI,  float* __restrict__ out) {
    __shared__ float red0[256], red1[256], red2[256];
    const int tid = threadIdx.x;
    float a1 = 0.f, a2 = 0.f, a3 = 0.f;

    for (int i = tid; i < BB; i += 256) {
        float s1 = 0.f, s2 = 0.f;
#pragma unroll
        for (int c = 0; c < NCHUNK; ++c) { s1 += S1a[c * BB + i]; s2 += S2a[c * BB + i]; }
        a1 += loss_term(s1, s2, p1[i], (float)(BB - 1));
        s1 = 0.f; s2 = 0.f;
#pragma unroll
        for (int c = 0; c < NCHUNK; ++c) { s1 += S1b[c * BB + i]; s2 += S2b[c * BB + i]; }
        a2 += loss_term(s1, s2, p2[i], (float)(BB - 1));
    }
    for (int i = tid; i < 2 * BB; i += 256) {
        float s1 = 0.f, s2 = 0.f;
#pragma unroll
        for (int c = 0; c < NCHUNK; ++c) { s1 += S1c[c * 2 * BB + i]; s2 += S2c[c * 2 * BB + i]; }
        a3 += loss_term(s1, s2, pI[i & (BB - 1)], (float)(2 * BB - 2));
    }

    red0[tid] = a1; red1[tid] = a2; red2[tid] = a3;
    __syncthreads();
    for (int s = 128; s > 0; s >>= 1) {
        if (tid < s) {
            red0[tid] += red0[tid + s];
            red1[tid] += red1[tid + s];
            red2[tid] += red2[tid + s];
        }
        __syncthreads();
    }
    if (tid == 0) {
        const float Linter = 0.5f * (red0[0] + red1[0]) * (1.0f / (float)BB);
        const float Lintra = red2[0] * (1.0f / (float)(2 * BB));
        out[0] = Linter + Lintra;
    }
}

// ---------------------------------------------------------------------------
// Launch
// ---------------------------------------------------------------------------
extern "C" void kernel_launch(void* const* d_in, const int* in_sizes, int n_in,
                              void* d_out, int out_size, void* d_ws, size_t ws_size,
                              hipStream_t stream) {
    const float* img1 = (const float*)d_in[0];
    const float* img2 = (const float*)d_in[1];
    // d_in[2] == text_z is unused by the reference loss
    const float* attr = (const float*)d_in[3];
    const int*   idx  = (const int*)d_in[4];
    float*       out  = (float*)d_out;

    // ---- workspace layout ----
    char* ws = (char*)d_ws;
    const size_t SB = (size_t)BB * DD * sizeof(unsigned short);   // 1 MB per plane
    unsigned short* G1hi = (unsigned short*)(ws + 0 * SB);
    unsigned short* G1lo = (unsigned short*)(ws + 1 * SB);
    unsigned short* G2hi = (unsigned short*)(ws + 2 * SB);
    unsigned short* G2lo = (unsigned short*)(ws + 3 * SB);
    unsigned short* AThi = (unsigned short*)(ws + 4 * SB);
    unsigned short* ATlo = (unsigned short*)(ws + 5 * SB);
    unsigned short* Chi  = (unsigned short*)(ws + 6 * SB);        // 2*SB
    unsigned short* Clo  = (unsigned short*)(ws + 8 * SB);        // 2*SB

    float* F = (float*)(ws + 10 * SB);
    float* S1a = F;                         // NCHUNK * 4096
    float* S2a = S1a + NCHUNK * BB;
    float* S1b = S2a + NCHUNK * BB;
    float* S2b = S1b + NCHUNK * BB;
    float* S1c = S2b + NCHUNK * BB;         // NCHUNK * 8192
    float* S2c = S1c + NCHUNK * 2 * BB;
    float* p1  = S2c + NCHUNK * 2 * BB;
    float* p2  = p1 + BB;
    float* pI  = p2 + BB;

    // 1) split planes
    prep_kernel<<<(BB * DD + 255) / 256, 256, 0, stream>>>(
        img1, img2, attr, idx, G1hi, G1lo, G2hi, G2lo, AThi, ATlo, Chi, Clo);

    // 2) pos terms (full fp32)
    pos_kernel<<<(BB + 255) / 256, 256, 0, stream>>>(img1, img2, attr, idx, p1, p2, pI);

    // 3) WMMA row sums (one block = 8 row tiles x 1 column chunk)
    //    inter 1: img1[idx] @ attr[idx]^T, 4096x4096, mask i==j
    {
        const int nRows = BB, chunkTiles = (BB / 16) / NCHUNK;
        const int blocks = (nRows / 16 / WAVES_PER_WG) * NCHUNK;
        gemm_rowsum_kernel<<<blocks, 256, 0, stream>>>(
            G1hi, G1lo, AThi, ATlo, S1a, S2a, nRows, chunkTiles, 0x7FFFFFFF);
    }
    //    inter 2: img2[idx] @ attr[idx]^T
    {
        const int nRows = BB, chunkTiles = (BB / 16) / NCHUNK;
        const int blocks = (nRows / 16 / WAVES_PER_WG) * NCHUNK;
        gemm_rowsum_kernel<<<blocks, 256, 0, stream>>>(
            G2hi, G2lo, AThi, ATlo, S1b, S2b, nRows, chunkTiles, 0x7FFFFFFF);
    }
    //    intra: concat @ concat^T, 8192x8192, mask (i % B) == (j % B)
    {
        const int nRows = 2 * BB, chunkTiles = (2 * BB / 16) / NCHUNK;
        const int blocks = (nRows / 16 / WAVES_PER_WG) * NCHUNK;
        gemm_rowsum_kernel<<<blocks, 256, 0, stream>>>(
            Chi, Clo, Chi, Clo, S1c, S2c, nRows, chunkTiles, BB - 1);
    }

    // 4) finalize to scalar
    finalize_kernel<<<1, 256, 0, stream>>>(S1a, S2a, S1b, S2b, S1c, S2c, p1, p2, pI, out);
}
